// GcnEdgeConvNet2_31593779430170
// MI455X (gfx1250) — compile-verified, matrix operand order checked by reading, not run on hardware
//
#include <hip/hip_runtime.h>

// ---------------------------------------------------------------------------
// GCN + edge MLP for MI455X (gfx1250, wave32).
// Node-conv phase: memory/atomic bound -> scalar kernels.
// Edge MLP phase (3.2M edges x (80->40->40->2)): bf16 WMMA 16x16x32 with
// double-buffered async global->LDS row gather (ASYNCcnt pipelined).
// ---------------------------------------------------------------------------

typedef __attribute__((ext_vector_type(16))) __bf16 v16bf;
typedef __attribute__((ext_vector_type(8)))  float  v8f;

__device__ __forceinline__ unsigned short f2bf(float x) {
    unsigned int u = __float_as_uint(x);
    u += 0x7FFFu + ((u >> 16) & 1u);          // round-to-nearest-even
    return (unsigned short)(u >> 16);
}
__device__ __forceinline__ __bf16 us2bf(unsigned short u) {
    union { unsigned short s; __bf16 b; } c; c.s = u; return c.b;
}
// 16-bit operand layout (ISA 7.12.2): lane<16 holds K 0..7 / 16..23,
// lane>=16 holds K 8..15 / 24..31; half-pairs are consecutive K.
__device__ __forceinline__ int kmap(int lane, int h) {
    return (((lane >> 4) & 1) << 3) + (h & 7) + ((h >> 3) << 4);
}

// ------------------------------ utility kernels ----------------------------

__global__ void zero_f32_k(float* __restrict__ p, long long n) {
    long long i = (long long)blockIdx.x * blockDim.x + threadIdx.x;
    if (i < n) p[i] = 0.f;
}

__global__ void degree_k(const long long* __restrict__ dst, float* __restrict__ deg,
                         long long nE) {
    long long e = (long long)blockIdx.x * blockDim.x + threadIdx.x;
    if (e < nE) atomicAdd(&deg[dst[e]], 1.0f);
}

__global__ void invdeg_k(float* __restrict__ deg, int nN) {
    int i = blockIdx.x * blockDim.x + threadIdx.x;
    if (i < nN) deg[i] = 1.0f / fmaxf(deg[i], 1.0f);
}

__global__ void pad_bias_k(const float* __restrict__ b, float* __restrict__ bp,
                           int n, int npad) {
    int i = threadIdx.x;
    if (i < npad) bp[i] = (i < n) ? b[i] : 0.f;
}

// Pack W [Kin x Nout] row-major into WMMA B fragments, zero-padded to
// KC*32 x NT*16.  frag[((kc*NT+nt)*32 + lane)*16 + h] = W[k][n].
__global__ void prep_frag_k(const float* __restrict__ W, unsigned short* __restrict__ frag,
                            int Kin, int Nout, int KC, int NT) {
    int t = blockIdx.x * blockDim.x + threadIdx.x;
    int total = KC * NT * 32 * 16;
    if (t >= total) return;
    int h    = t & 15;
    int lane = (t >> 4) & 31;
    int f    = t >> 9;
    int kc = f / NT, nt = f % NT;
    int k = kc * 32 + kmap(lane, h);
    int n = nt * 16 + (lane & 15);
    float v = (k < Kin && n < Nout) ? W[k * Nout + n] : 0.f;
    frag[t] = f2bf(v);
}

// ------------------------------ node-conv phase ----------------------------

// agg[dst[e]][f] += h[src[e]][f]   (f = blockIdx.y)
__global__ void scatter_k(const float* __restrict__ h,
                          const long long* __restrict__ src,
                          const long long* __restrict__ dst,
                          float* __restrict__ agg, long long nE, int d) {
    long long e = (long long)blockIdx.x * blockDim.x + threadIdx.x;
    int f = blockIdx.y;
    if (e < nE)
        atomicAdd(&agg[dst[e] * d + f], h[src[e] * d + f]);
}

// out[node][j] = relu( concat(h[node], agg[node]*rdeg[node]) . W[:,j] + b[j] )
__global__ void node_linear_k(const float* __restrict__ h,
                              const float* __restrict__ agg,
                              const float* __restrict__ rdeg,
                              const float* __restrict__ W,
                              const float* __restrict__ b,
                              float* __restrict__ out,
                              int nN, int din, int dout) {
    int node = blockIdx.x * blockDim.x + threadIdx.x;
    int j = blockIdx.y;
    if (node >= nN) return;
    float acc = b[j];
    float r = rdeg[node];
    const float* hp = h + (long long)node * din;
    const float* ap = agg + (long long)node * din;
    for (int f = 0; f < din; ++f) acc += hp[f] * W[f * dout + j];
    for (int f = 0; f < din; ++f) acc += (ap[f] * r) * W[(din + f) * dout + j];
    out[(long long)node * dout + j] = fmaxf(acc, 0.f);
}

__global__ void f2bf_k(const float* __restrict__ h, unsigned short* __restrict__ hbf,
                       long long n) {
    long long i = (long long)blockIdx.x * blockDim.x + threadIdx.x;
    if (i < n) hbf[i] = f2bf(h[i]);
}

// ------------------------------ edge MLP (WMMA) ----------------------------
// One wave = 16 edges per tile (M=16), persistent grid-stride over tiles with
// double-buffered async global->LDS gather:
//   lane l copies one 80B node-feature row (edge = lane&15, src/dst = lane>>4)
//   with 5 x global_load_async_to_lds_b128 (ASYNCcnt).
// Layer1: K=96 (3 chunks) x N=48 (3 tiles) -> 9 WMMA
// Layer2: K=64 (2)        x N=48 (3)       -> 6 WMMA
// Layer3: K=64 (2)        x N=16 (1)       -> 2 WMMA
__global__ __launch_bounds__(128) void edge_mlp_k(
        const unsigned short* __restrict__ hbf,
        const long long* __restrict__ src,
        const long long* __restrict__ dst,
        const unsigned short* __restrict__ w7f,
        const unsigned short* __restrict__ w8f,
        const unsigned short* __restrict__ w9f,
        const float* __restrict__ b7p,
        const float* __restrict__ b8p,
        const float* __restrict__ b9p,
        float* __restrict__ out, long long nE)
{
    // stage at LDS offset 0: [wave][buf][row 0..15][96 bf16]; row stride 192B
    __shared__ __align__(16) unsigned short stage[4][2][16][96];
    const int lane = threadIdx.x & 31;
    const int wave = threadIdx.x >> 5;
    const long long nTiles   = (nE + 15) >> 4;
    const long long waveId   = (long long)blockIdx.x * 4 + wave;
    const long long numWaves = (long long)gridDim.x * 4;
    if (waveId >= nTiles) return;             // wave-uniform, no block barriers used

    const int sm = lane & 15;                 // staging: edge row within tile
    const int sw = lane >> 4;                 // staging: 0 = src row, 1 = dst row
    const int m0 = (lane >> 4) << 3;          // C/D layout: vgpr j -> row m0+j
    const int n0 = lane & 15;                 //             col n0
    const v8f vzero = {0.f, 0.f, 0.f, 0.f, 0.f, 0.f, 0.f, 0.f};

    // Zero the fixed K-pad columns 80..95 of both buffers once (never
    // overwritten: async writes cols 0..79, inter-layer writes cols 0..63).
    {
        unsigned int* rb = (unsigned int*)&stage[wave][0][0][0];
        #pragma unroll
        for (int i = 0; i < 8; ++i) {
            int w = lane + 32 * i;            // 0..255: 2 bufs x 16 rows x 8 dwords
            int b = w >> 7;
            int r = (w >> 3) & 15;
            int c = w & 7;
            rb[(b * 16 + r) * 48 + 40 + c] = 0u;
        }
    }
    asm volatile("s_wait_dscnt 0x0" ::: "memory");

    // Async gather of one tile into buffer `bsel`: each lane copies one 80B
    // row: LDS[ldsoff+imm] = MEM[saddr + vaddr + imm].  Tail edges clamp the
    // address (EXEC must stay all-ones for the WMMAs).
    auto issue_stage = [&](long long tile, int bsel) {
        long long e = tile * 16 + sm; if (e >= nE) e = nE - 1;
        long long node = sw ? dst[e] : src[e];
        unsigned vaddr  = (unsigned)(node * 80ll);
        unsigned ldsoff = (unsigned)((((wave * 2 + bsel) * 16 + sm) * 192) + sw * 80);
        asm volatile(
            "global_load_async_to_lds_b128 %0, %1, %2 offset:0\n\t"
            "global_load_async_to_lds_b128 %0, %1, %2 offset:16\n\t"
            "global_load_async_to_lds_b128 %0, %1, %2 offset:32\n\t"
            "global_load_async_to_lds_b128 %0, %1, %2 offset:48\n\t"
            "global_load_async_to_lds_b128 %0, %1, %2 offset:64"
            :: "v"(ldsoff), "v"(vaddr), "s"(hbf) : "memory");
    };

    auto loadA = [&](int bsel, int kc) {
        v16bf a;
        #pragma unroll
        for (int h = 0; h < 16; ++h)
            a[h] = us2bf(stage[wave][bsel][sm][kc * 32 + kmap(lane, h)]);
        return a;
    };

    issue_stage(waveId, 0);
    int buf = 0;
    for (long long tile = waveId; tile < nTiles; tile += numWaves, buf ^= 1) {
        // prefetch next tile into the other buffer, then wait for current:
        // 5 outstanding -> current retired (async loads retire in order)
        long long nxt = tile + numWaves;
        if (nxt < nTiles) {
            issue_stage(nxt, buf ^ 1);
            asm volatile("s_wait_asynccnt 0x5" ::: "memory");
        } else {
            asm volatile("s_wait_asynccnt 0x0" ::: "memory");
        }

        unsigned int* rb32 = (unsigned int*)&stage[wave][buf][0][0];

        // ---- layer 1: [16x96] @ [96x48]
        v8f a1[3] = {vzero, vzero, vzero};
        #pragma unroll
        for (int kc = 0; kc < 3; ++kc) {
            v16bf a = loadA(buf, kc);
            #pragma unroll
            for (int nt = 0; nt < 3; ++nt) {
                v16bf b = *(const v16bf*)(w7f + (((kc * 3 + nt) * 32 + lane) << 4));
                a1[nt] = __builtin_amdgcn_wmma_f32_16x16x32_bf16(
                             false, a, false, b, (short)0, a1[nt], false, false);
            }
        }
        #pragma unroll
        for (int nt = 0; nt < 3; ++nt) {
            float bias = b7p[nt * 16 + n0];
            #pragma unroll
            for (int j = 0; j < 8; ++j)
                stage[wave][buf][m0 + j][nt * 16 + n0] =
                    f2bf(fmaxf(a1[nt][j] + bias, 0.f));
        }
        // re-zero cols 48..63 (clobbered each tile by the dst-row async copy)
        #pragma unroll
        for (int i = 0; i < 4; ++i) {
            int w = lane + 32 * i;               // 0..127: 16 rows x 8 dwords
            rb32[(w >> 3) * 48 + 24 + (w & 7)] = 0u;
        }
        asm volatile("s_wait_dscnt 0x0" ::: "memory");

        // ---- layer 2: [16x64] @ [64x48]
        v8f a2[3] = {vzero, vzero, vzero};
        #pragma unroll
        for (int kc = 0; kc < 2; ++kc) {
            v16bf a = loadA(buf, kc);
            #pragma unroll
            for (int nt = 0; nt < 3; ++nt) {
                v16bf b = *(const v16bf*)(w8f + (((kc * 3 + nt) * 32 + lane) << 4));
                a2[nt] = __builtin_amdgcn_wmma_f32_16x16x32_bf16(
                             false, a, false, b, (short)0, a2[nt], false, false);
            }
        }
        #pragma unroll
        for (int nt = 0; nt < 3; ++nt) {
            float bias = b8p[nt * 16 + n0];
            #pragma unroll
            for (int j = 0; j < 8; ++j)
                stage[wave][buf][m0 + j][nt * 16 + n0] =
                    f2bf(fmaxf(a2[nt][j] + bias, 0.f));
        }
        asm volatile("s_wait_dscnt 0x0" ::: "memory");

        // ---- layer 3: [16x64] @ [64x16]
        v8f a3 = vzero;
        #pragma unroll
        for (int kc = 0; kc < 2; ++kc) {
            v16bf a = loadA(buf, kc);
            v16bf b = *(const v16bf*)(w9f + ((kc * 32 + lane) << 4));
            a3 = __builtin_amdgcn_wmma_f32_16x16x32_bf16(
                     false, a, false, b, (short)0, a3, false, false);
        }

        // ---- bias + relu + softmax over the 2 classes (lanes n0=0/1 pair)
        float bias9 = b9p[n0];
        #pragma unroll
        for (int j = 0; j < 8; ++j) {
            float v = fmaxf(a3[j] + bias9, 0.f);
            float o = __shfl_xor(v, 1, 32);
            if (n0 < 2) {
                float mx = fmaxf(v, o);
                float e0 = __expf(v - mx);
                float e1 = __expf(o - mx);
                long long e = tile * 16 + m0 + j;
                if (e < nE) out[e * 2 + n0] = e0 / (e0 + e1);
            }
        }
    }
}

// ------------------------------ workspace layout ---------------------------

static constexpr size_t OFF_RDEG = 0;                         // 100K f32
static constexpr size_t OFF_HA   = 1ull  << 20;               // 100K x 40 f32
static constexpr size_t OFF_HB   = OFF_HA  + (18ull << 20);
static constexpr size_t OFF_AGG  = OFF_HB  + (18ull << 20);   // agg; later reused as hbf
static constexpr size_t OFF_W7F  = OFF_AGG + (18ull << 20);
static constexpr size_t OFF_W8F  = OFF_W7F + (16u << 10);
static constexpr size_t OFF_W9F  = OFF_W8F + (16u << 10);
static constexpr size_t OFF_B7   = OFF_W9F + (16u << 10);
static constexpr size_t OFF_B8   = OFF_B7 + 256;
static constexpr size_t OFF_B9   = OFF_B8 + 256;

extern "C" void kernel_launch(void* const* d_in, const int* in_sizes, int n_in,
                              void* d_out, int out_size, void* d_ws, size_t ws_size,
                              hipStream_t stream) {
    const float*     x    = (const float*)d_in[0];
    const long long* eidx = (const long long*)d_in[2];   // edge_index is int64
    const long long  nE   = in_sizes[2] / 2;
    const int        nN   = in_sizes[0] / 16;
    const long long* src  = eidx;
    const long long* dst  = eidx + nE;

    const float* W[9]; const float* B[9];
    for (int i = 0; i < 9; ++i) {
        W[i] = (const float*)d_in[3 + 2 * i];
        B[i] = (const float*)d_in[4 + 2 * i];
    }

    char* ws = (char*)d_ws;
    float*          rdeg = (float*)(ws + OFF_RDEG);
    float*          hA   = (float*)(ws + OFF_HA);
    float*          hB   = (float*)(ws + OFF_HB);
    float*          agg  = (float*)(ws + OFF_AGG);
    unsigned short* hbf  = (unsigned short*)(ws + OFF_AGG);  // reuse agg after node phase
    unsigned short* w7f  = (unsigned short*)(ws + OFF_W7F);
    unsigned short* w8f  = (unsigned short*)(ws + OFF_W8F);
    unsigned short* w9f  = (unsigned short*)(ws + OFF_W9F);
    float*          b7p  = (float*)(ws + OFF_B7);
    float*          b8p  = (float*)(ws + OFF_B8);
    float*          b9p  = (float*)(ws + OFF_B9);

    // degree -> 1/max(deg,1)
    zero_f32_k<<<(nN + 255) / 256, 256, 0, stream>>>(rdeg, nN);
    degree_k<<<(int)((nE + 255) / 256), 256, 0, stream>>>(dst, rdeg, nE);
    invdeg_k<<<(nN + 255) / 256, 256, 0, stream>>>(rdeg, nN);

    // weight fragments + padded biases for the WMMA edge MLP
    prep_frag_k<<<(9 * 512 + 255) / 256, 256, 0, stream>>>(W[6], w7f, 80, 40, 3, 3);
    prep_frag_k<<<(6 * 512 + 255) / 256, 256, 0, stream>>>(W[7], w8f, 40, 40, 2, 3);
    prep_frag_k<<<(2 * 512 + 255) / 256, 256, 0, stream>>>(W[8], w9f, 40,  2, 2, 1);
    pad_bias_k<<<1, 64, 0, stream>>>(B[6], b7p, 40, 48);
    pad_bias_k<<<1, 64, 0, stream>>>(B[7], b8p, 40, 48);
    pad_bias_k<<<1, 64, 0, stream>>>(B[8], b9p,  2, 16);

    // six NodeConv layers
    const int din[6] = {16, 15, 25, 30, 30, 40};
    const int dou[6] = {15, 25, 30, 30, 40, 40};
    const float* cur = x;
    for (int l = 0; l < 6; ++l) {
        float* outb = (l & 1) ? hB : hA;
        long long na = (long long)nN * din[l];
        zero_f32_k<<<(int)((na + 255) / 256), 256, 0, stream>>>(agg, na);
        dim3 sg((unsigned)((nE + 255) / 256), (unsigned)din[l]);
        scatter_k<<<sg, 256, 0, stream>>>(cur, src, dst, agg, nE, din[l]);
        dim3 lg((unsigned)((nN + 255) / 256), (unsigned)dou[l]);
        node_linear_k<<<lg, 256, 0, stream>>>(cur, agg, rdeg, W[l], B[l], outb,
                                              nN, din[l], dou[l]);
        cur = outb;
    }

    // final node features -> bf16 (halves edge-gather bytes, feeds WMMA)
    long long nh = (long long)nN * 40;
    f2bf_k<<<(int)((nh + 255) / 256), 256, 0, stream>>>(cur, hbf, nh);

    // edge MLP: persistent waves, 16 edges per tile, double-buffered async gather
    long long nTiles = (nE + 15) / 16;
    long long blocksNeeded = (nTiles + 3) / 4;
    int blocks = (int)(blocksNeeded < 4096 ? blocksNeeded : 4096);
    edge_mlp_k<<<blocks, 128, 0, stream>>>(
        hbf, src, dst, w7f, w8f, w9f, b7p, b8p, b9p, (float*)d_out, nE);
}
